// GPT_80994493268534
// MI455X (gfx1250) — compile-verified, matrix operand order checked by reading, not run on hardware
//
#include <hip/hip_runtime.h>
#include <hip/hip_bf16.h>
#include <math.h>

// ---------------------------------------------------------------------------
// GPT forward (B=8,T=1024,C=512,L=6,H=8,DH=64,OD=768,FF=2048) on gfx1250.
// bf16 WMMA, fp32 residual/softmax/LN.
//   - GEMM tiles staged by the Tensor Data Mover (tensor_load_to_lds,
//     TENSORcnt), double buffered; D# pad fields reproduce the padded LDS rows.
//   - Attention K/V staged by per-lane global_load_async_to_lds_b128
//     (ASYNCcnt), double buffered.
//   - Weights pre-transposed to [N][K] bf16; V written transposed [B][C][T]
//     so every WMMA fragment read is a contiguous pair of ds_load_b128.
// ---------------------------------------------------------------------------

#define B_DIM  8
#define T_DIM  1024
#define C_DIM  512
#define H_DIM  8
#define DH_DIM 64
#define OD_DIM 768
#define FF_DIM 2048
#define NTOK   (B_DIM * T_DIM)   // 8192

typedef __attribute__((ext_vector_type(16))) __bf16 v16bf;
typedef __attribute__((ext_vector_type(8)))  __bf16 v8bf;
typedef __attribute__((ext_vector_type(8)))  float  v8f;
typedef __attribute__((ext_vector_type(4)))  unsigned int v4u;
typedef __attribute__((ext_vector_type(4)))  int v4i;
typedef __attribute__((ext_vector_type(8)))  int v8i;

__device__ __forceinline__ v8f wmma_bf16(v16bf a, v16bf b, v8f c) {
  return __builtin_amdgcn_wmma_f32_16x16x32_bf16(false, a, false, b, (short)0, c,
                                                 false, false);
}

// ---- per-lane async 16B global->LDS copy (ASYNCcnt) ----
__device__ __forceinline__ void async_b128(void* lds, const void* gbase,
                                           int byteoff) {
  asm volatile("global_load_async_to_lds_b128 %0, %1, %2 offset:0"
               :
               : "v"((unsigned)(uintptr_t)lds), "v"(byteoff), "s"(gbase)
               : "memory");
}
__device__ __forceinline__ void wait_async0() {
  asm volatile("s_wait_asynccnt 0x0" ::: "memory");
}

// ---- Tensor Data Mover: 2D bf16 tile -> padded LDS rows (TENSORcnt) ----
// tile_w elements per row (64 -> 128B), tile_h rows, global row stride in
// elements. LDS rows padded by 16B after every 128B: matches stride 72 bf16.
__device__ __forceinline__ void tdm_load_2d(unsigned lds_addr, const void* gptr,
                                            unsigned tile_w, unsigned tile_h,
                                            unsigned row_stride_elems) {
  const unsigned long long ga = (unsigned long long)(uintptr_t)gptr;
  v4u g0;
  g0[0] = 1u;                                   // count=1 (valid descriptor)
  g0[1] = lds_addr;                             // LDS byte address
  g0[2] = (unsigned)ga;                         // global_addr[31:0]
  g0[3] = (unsigned)((ga >> 32) & 0x01ffffffu)  // global_addr[56:32]
          | (2u << 30);                         // type = 2 ("image")
  const unsigned td0 = 1u << 24, td1 = 1u << 24;  // huge dims: no OOB clip
  const unsigned long long s0 = row_stride_elems; // dim0 stride (elements)
  v8i g1;
  g1[0] = (int)((1u << 16)      // data_size = 1 (2 bytes)
              | (1u << 20)      // pad_enable
              | (4u << 22)      // pad_interval: 32 DWORDs (128B)
              | (3u << 25));    // pad_amount:   4 DWORDs (16B)
  g1[1] = (int)((td0 & 0xffffu) << 16);                    // td0[15:0]
  g1[2] = (int)((td0 >> 16) | ((td1 & 0xffffu) << 16));    // td0 hi | td1 lo
  g1[3] = (int)((td1 >> 16) | (tile_w << 16));             // td1 hi | tile_dim0
  g1[4] = (int)tile_h;                                     // tile_dim1 (dim2=0)
  g1[5] = (int)(unsigned)(s0 & 0xffffffffu);               // stride0 lo
  g1[6] = (int)(unsigned)((s0 >> 32) & 0xffffu);           // stride0 hi
  g1[7] = 0;
  v4i g2 = {0, 0, 0, 0};
  v4i g3 = {0, 0, 0, 0};
#if __clang_major__ >= 23
  v8i g4 = {0, 0, 0, 0, 0, 0, 0, 0};
  __builtin_amdgcn_tensor_load_to_lds(g0, g1, g2, g3, g4, 0);
#else
  __builtin_amdgcn_tensor_load_to_lds(g0, g1, g2, g3, 0);
#endif
}
__device__ __forceinline__ void wait_tensor0() {
  __builtin_amdgcn_s_wait_tensorcnt(0);
}

// ISA 7.12.2 16-bit operand element->K map.
__device__ __forceinline__ int kmap(int i, int hi) {
  return i + ((i >= 8) ? 8 : 0) + hi * 8;
}

// Fragment from a row-major LDS tile: lane -> row, elements -> K (kmap order).
__device__ __forceinline__ v16bf frag_row(const __bf16* p, int stride, int row0,
                                          int kofs, int lane) {
  const int hi = lane >> 4;
  const int m  = row0 + (lane & 15);
  v16bf f;
#pragma unroll
  for (int i = 0; i < 16; ++i) f[i] = p[m * stride + kmap(i, hi) + kofs];
  return f;
}

__device__ __forceinline__ float gelu_exact(float x) {
  return 0.5f * x * (1.0f + erff(x * 0.70710678118654752f));
}

// ---------------------------------------------------------------------------
// Weight prep: out[n*K+k] = (bf16)in[k*N+n]   (tiled 32x32 transpose)
// ---------------------------------------------------------------------------
__global__ __launch_bounds__(256) void transpose_cvt_bf16(
    const float* __restrict__ in, __bf16* __restrict__ out, int K, int N) {
  __shared__ float tile[32][33];
  const int k0 = blockIdx.y * 32, n0 = blockIdx.x * 32;
  const int tx = threadIdx.x & 31, ty = threadIdx.x >> 5;
#pragma unroll
  for (int i = 0; i < 32; i += 8)
    tile[ty + i][tx] = in[(size_t)(k0 + ty + i) * N + (n0 + tx)];
  __syncthreads();
#pragma unroll
  for (int i = 0; i < 32; i += 8)
    out[(size_t)(n0 + ty + i) * K + (k0 + tx)] = (__bf16)tile[tx][ty + i];
}

__global__ __launch_bounds__(256) void cvt_bf16_kernel(
    const float* __restrict__ in, __bf16* __restrict__ out, int n) {
  int i = (blockIdx.x * 256 + threadIdx.x) * 4;
  if (i + 3 < n) {
    const float4 v = *(const float4*)(in + i);
    out[i + 0] = (__bf16)v.x;
    out[i + 1] = (__bf16)v.y;
    out[i + 2] = (__bf16)v.z;
    out[i + 3] = (__bf16)v.w;
  }
}

// ---------------------------------------------------------------------------
// GEMM: out = epilogue(A[M,K](bf16) @ BT[N,K](bf16)^T + bias)
//  MODE 0: bf16 out[M,N] = acc + bias
//  MODE 1: bf16 out[M,N] = gelu(acc + bias)
//  MODE 2: f32  out[M,N] = acc + bias + extra[row*N+col]           (residual)
//  MODE 3: f32  out[M,N] = gelu(acc + bias) + extra[(row%T)*N+col] (pos emb)
//  MODE 4: bf16 out[(b*C+col)*T + t] = acc + bias   (transposed V, b128 pack)
// 128 threads / 4 waves; block tile 128x64, wave 32x64, K-step 64.
// Tiles staged by TDM (wave0: A, wave1: B), double buffered.
// ---------------------------------------------------------------------------
template <int MODE>
__global__ __launch_bounds__(128) void gemm_bf16_async(
    const __bf16* __restrict__ A, const __bf16* __restrict__ BT,
    const float* __restrict__ bias, const float* __restrict__ extra,
    void* __restrict__ outp, int M, int N, int K) {
  constexpr int TM = 128, TN = 64, TK = 64, SA = 72;  // 144B rows, 16B aligned
  __shared__ __bf16 As[2][TM * SA];
  __shared__ __bf16 Bs[2][TN * SA];

  const int tid  = threadIdx.x;
  const int lane = tid & 31;
  const int wave = tid >> 5;
  const int wm   = wave * 32;
  const int m0   = blockIdx.y * TM;
  const int n0   = blockIdx.x * TN;

  auto issue = [&](int buf, int k0) {
    if (wave == 0)
      tdm_load_2d((unsigned)(uintptr_t)&As[buf][0], A + (size_t)m0 * K + k0,
                  TK, TM, K);
    else if (wave == 1)
      tdm_load_2d((unsigned)(uintptr_t)&Bs[buf][0], BT + (size_t)n0 * K + k0,
                  TK, TN, K);
  };

  v8f acc[2][4];
#pragma unroll
  for (int i = 0; i < 2; ++i)
#pragma unroll
    for (int j = 0; j < 4; ++j)
#pragma unroll
      for (int e = 0; e < 8; ++e) acc[i][j][e] = 0.0f;

  issue(0, 0);
  const int nk = K / TK;
  for (int kk = 0; kk < nk; ++kk) {
    const int cur = kk & 1;
    wait_tensor0();
    __syncthreads();
    if (kk + 1 < nk) issue(cur ^ 1, (kk + 1) * TK);
#pragma unroll
    for (int ks = 0; ks < 2; ++ks) {
      v16bf af[2], bf2[4];
#pragma unroll
      for (int i = 0; i < 2; ++i)
        af[i] = frag_row(&As[cur][0], SA, wm + i * 16, ks * 32, lane);
#pragma unroll
      for (int j = 0; j < 4; ++j)
        bf2[j] = frag_row(&Bs[cur][0], SA, j * 16, ks * 32, lane);
#pragma unroll
      for (int i = 0; i < 2; ++i)
#pragma unroll
        for (int j = 0; j < 4; ++j)
          acc[i][j] = wmma_bf16(af[i], bf2[j], acc[i][j]);
    }
    __syncthreads();
  }

  const int hi = lane >> 4;
  const int nl = lane & 15;
#pragma unroll
  for (int i = 0; i < 2; ++i)
#pragma unroll
    for (int j = 0; j < 4; ++j) {
      const int col = n0 + j * 16 + nl;
      if (MODE == 4) {
        // 8 accumulator elements == 8 consecutive tokens -> one b128 store
        const int t0   = m0 + wm + i * 16 + hi * 8;
        const int bidx = t0 >> 10;       // / T_DIM
        const int tloc = t0 & (T_DIM - 1);
        v8bf pk;
#pragma unroll
        for (int r = 0; r < 8; ++r) pk[r] = (__bf16)(acc[i][j][r] + bias[col]);
        *(v8bf*)&((__bf16*)outp)[((size_t)bidx * C_DIM + col) * T_DIM + tloc] = pk;
      } else {
#pragma unroll
        for (int r = 0; r < 8; ++r) {
          const int row = m0 + wm + i * 16 + r + hi * 8;
          float v = acc[i][j][r] + bias[col];
          if (MODE == 1) v = gelu_exact(v);
          if (MODE == 2) v += extra[(size_t)row * N + col];
          if (MODE == 3)
            v = gelu_exact(v) + extra[(size_t)(row & (T_DIM - 1)) * N + col];
          if (MODE == 0 || MODE == 1)
            ((__bf16*)outp)[(size_t)row * N + col] = (__bf16)v;
          else
            ((float*)outp)[(size_t)row * N + col] = v;
        }
      }
    }
}

// ---------------------------------------------------------------------------
// Causal flash attention: block = (b, h, 64 query rows); 4 waves x 16 rows.
// Q/K from [B,T,C] bf16; V from transposed [B,C,T] bf16. Diagonal block
// peeled; K/V double-buffered via per-lane async copies (ASYNCcnt).
// ---------------------------------------------------------------------------
__global__ __launch_bounds__(128) void attn_flash(const __bf16* __restrict__ Qg,
                                                  const __bf16* __restrict__ Kg,
                                                  const __bf16* __restrict__ Vt,
                                                  __bf16* __restrict__ Yg) {
  constexpr int LS = DH_DIM + 8;  // 72
  __shared__ __bf16 Qs[64 * LS];
  __shared__ __bf16 Ks[2][64 * LS];      // [key][dh]
  __shared__ __bf16 Vs[2][64 * LS];      // [dh][key]  (from transposed V)
  __shared__ __bf16 Ps[4][16 * LS];

  const int tid  = threadIdx.x;
  const int lane = tid & 31;
  const int wave = tid >> 5;
  const int hi   = lane >> 4;
  const int nl   = lane & 15;
  const int qb   = blockIdx.x;
  const int bh   = blockIdx.y;
  const int b    = bh >> 3;
  const int h    = bh & 7;
  const int hoff = h * DH_DIM;
  const size_t base   = (size_t)b * T_DIM * C_DIM;           // [B,T,C] bf16
  const size_t vbase0 = ((size_t)b * C_DIM + hoff) * T_DIM;  // [B,C,T] bf16

  const int tr = tid >> 3;
  const int tc = tid & 7;

  {
    const __bf16* qbase = Qg + base + (size_t)(qb * 64) * C_DIM + hoff;
#pragma unroll
    for (int it = 0; it < 4; ++it) {
      int r = tr + it * 16;
      async_b128(&Qs[r * LS + tc * 8], qbase, r * (C_DIM * 2) + tc * 16);
    }
  }
  auto issue_kv = [&](int buf, int kb) {
    const __bf16* kbase = Kg + base + (size_t)(kb * 64) * C_DIM + hoff;
    const __bf16* vbase = Vt + vbase0 + kb * 64;
#pragma unroll
    for (int it = 0; it < 4; ++it) {
      int r = tr + it * 16;
      async_b128(&Ks[buf][r * LS + tc * 8], kbase, r * (C_DIM * 2) + tc * 16);
      async_b128(&Vs[buf][r * LS + tc * 8], vbase, r * (T_DIM * 2) + tc * 16);
    }
  };

  const int wq = wave * 16;
  float mrun[8], lrun[8];
  v8f accO[4];
#pragma unroll
  for (int r = 0; r < 8; ++r) { mrun[r] = -1e30f; lrun[r] = 0.0f; }
#pragma unroll
  for (int j = 0; j < 4; ++j)
#pragma unroll
    for (int e = 0; e < 8; ++e) accO[j][e] = 0.0f;

  const float scale = 0.125f;  // 1/sqrt(DH)

  auto process = [&](int cur, int kb, bool domask) {
    v8f s[4];
#pragma unroll
    for (int j = 0; j < 4; ++j)
#pragma unroll
      for (int e = 0; e < 8; ++e) s[j][e] = 0.0f;
#pragma unroll
    for (int ks = 0; ks < 2; ++ks) {
      v16bf aq = frag_row(Qs, LS, wq, ks * 32, lane);
#pragma unroll
      for (int j = 0; j < 4; ++j) {
        v16bf bk = frag_row(&Ks[cur][0], LS, j * 16, ks * 32, lane);
        s[j] = wmma_bf16(aq, bk, s[j]);
      }
    }
#pragma unroll
    for (int j = 0; j < 4; ++j)
#pragma unroll
      for (int r = 0; r < 8; ++r) {
        float v = s[j][r] * scale;
        if (domask) {
          int kgl = kb * 64 + j * 16 + nl;
          int qgl = qb * 64 + wq + r + hi * 8;
          v = (kgl > qgl) ? -1e30f : v;
        }
        s[j][r] = v;
      }

#pragma unroll
    for (int r = 0; r < 8; ++r) {
      float mloc = fmaxf(fmaxf(s[0][r], s[1][r]), fmaxf(s[2][r], s[3][r]));
#pragma unroll
      for (int off = 1; off < 16; off <<= 1)
        mloc = fmaxf(mloc, __shfl_xor(mloc, off, 32));
      float mnew  = fmaxf(mrun[r], mloc);
      float alpha = __expf(mrun[r] - mnew);
      float rs = 0.0f;
#pragma unroll
      for (int j = 0; j < 4; ++j) {
        float p = __expf(s[j][r] - mnew);
        rs += p;
        Ps[wave][(r + hi * 8) * LS + j * 16 + nl] = (__bf16)p;
      }
#pragma unroll
      for (int off = 1; off < 16; off <<= 1) rs += __shfl_xor(rs, off, 32);
      lrun[r] = lrun[r] * alpha + rs;
      mrun[r] = mnew;
#pragma unroll
      for (int j = 0; j < 4; ++j) accO[j][r] *= alpha;
    }

#pragma unroll
    for (int ks = 0; ks < 2; ++ks) {
      v16bf ap = frag_row(Ps[wave], LS, 0, ks * 32, lane);
#pragma unroll
      for (int j = 0; j < 4; ++j) {
        v16bf bv = frag_row(&Vs[cur][0], LS, j * 16, ks * 32, lane);
        accO[j] = wmma_bf16(ap, bv, accO[j]);
      }
    }
    __syncthreads();
  };

  issue_kv(0, 0);
  for (int kb = 0; kb < qb; ++kb) {   // interior: mask-free
    const int cur = kb & 1;
    wait_async0();
    __syncthreads();
    issue_kv(cur ^ 1, kb + 1);
    process(cur, kb, false);
  }
  {  // diagonal block
    const int cur = qb & 1;
    wait_async0();
    __syncthreads();
    process(cur, qb, true);
  }

#pragma unroll
  for (int j = 0; j < 4; ++j)
#pragma unroll
    for (int r = 0; r < 8; ++r) {
      int t = qb * 64 + wq + r + hi * 8;
      int c = hoff + j * 16 + nl;
      Yg[base + (size_t)t * C_DIM + c] = (__bf16)(accO[j][r] / lrun[r]);
    }
}

// ---------------------------------------------------------------------------
// LayerNorm over C=512; OB=1 -> bf16 out, OB=0 -> fp32 out.
// ---------------------------------------------------------------------------
template <int OB>
__global__ __launch_bounds__(256) void layernorm_512(
    const float* __restrict__ x, const float* __restrict__ w,
    const float* __restrict__ b, void* __restrict__ outp) {
  __shared__ float wsum[8];
  __shared__ float wvar[8];
  const int tid  = threadIdx.x;
  const int lane = tid & 31;
  const int wv   = tid >> 5;
  const float* xr = x + (size_t)blockIdx.x * C_DIM;
  float v0 = xr[tid], v1 = xr[tid + 256];

  float s = v0 + v1;
#pragma unroll
  for (int off = 16; off; off >>= 1) s += __shfl_xor(s, off, 32);
  if (lane == 0) wsum[wv] = s;
  __syncthreads();
  float tot = 0.0f;
#pragma unroll
  for (int i = 0; i < 8; ++i) tot += wsum[i];
  const float mu = tot * (1.0f / C_DIM);

  float d0 = v0 - mu, d1 = v1 - mu;
  float vs = d0 * d0 + d1 * d1;
#pragma unroll
  for (int off = 16; off; off >>= 1) vs += __shfl_xor(vs, off, 32);
  if (lane == 0) wvar[wv] = vs;
  __syncthreads();
  float vtot = 0.0f;
#pragma unroll
  for (int i = 0; i < 8; ++i) vtot += wvar[i];
  const float rstd = rsqrtf(vtot * (1.0f / C_DIM) + 1e-5f);

  float o0 = d0 * rstd * w[tid] + b[tid];
  float o1 = d1 * rstd * w[tid + 256] + b[tid + 256];
  if (OB) {
    __bf16* orow = (__bf16*)outp + (size_t)blockIdx.x * C_DIM;
    orow[tid] = (__bf16)o0;
    orow[tid + 256] = (__bf16)o1;
  } else {
    float* orow = (float*)outp + (size_t)blockIdx.x * C_DIM;
    orow[tid] = o0;
    orow[tid + 256] = o1;
  }
}

// ---------------------------------------------------------------------------
extern "C" void kernel_launch(void* const* d_in, const int* in_sizes, int n_in,
                              void* d_out, int out_size, void* d_ws,
                              size_t ws_size, hipStream_t stream) {
  (void)in_sizes; (void)n_in; (void)out_size; (void)ws_size;

  const float* obs     = (const float*)d_in[0];
  const float* obs_W   = (const float*)d_in[1];
  const float* obs_b   = (const float*)d_in[2];
  const float* pos_emb = (const float*)d_in[3];
  const float* ln1_w   = (const float*)d_in[4];
  const float* ln1_b   = (const float*)d_in[5];
  const float* Wq      = (const float*)d_in[6];
  const float* bq      = (const float*)d_in[7];
  const float* Wk      = (const float*)d_in[8];
  const float* bk      = (const float*)d_in[9];
  const float* Wv      = (const float*)d_in[10];
  const float* bv      = (const float*)d_in[11];
  const float* Wo      = (const float*)d_in[12];
  const float* bo      = (const float*)d_in[13];
  const float* ln2_w   = (const float*)d_in[14];
  const float* ln2_b   = (const float*)d_in[15];
  const float* W1      = (const float*)d_in[16];
  const float* b1      = (const float*)d_in[17];
  const float* W2      = (const float*)d_in[18];
  const float* b2      = (const float*)d_in[19];
  const float* lnf_w   = (const float*)d_in[20];
  const float* lnf_b   = (const float*)d_in[21];

  const size_t SEG = (size_t)NTOK * C_DIM;  // 4,194,304
  const size_t CC  = (size_t)C_DIM * C_DIM;
  const size_t CF  = (size_t)C_DIM * FF_DIM;
  const size_t LWSEG = 4 * CC + 2 * CF;

  float*  X  = (float*)d_ws;
  __bf16* bb = (__bf16*)(X + SEG);
  __bf16* Hb   = bb;
  __bf16* Qb   = Hb + SEG;
  __bf16* Kb   = Qb + SEG;
  __bf16* Vtb  = Kb + SEG;   // transposed V: [B][C][T]
  __bf16* Yb   = Vtb + SEG;
  __bf16* HID  = Qb;         // aliases Q..Y (NTOK * FF)
  __bf16* obsB = Yb + SEG;
  __bf16* oWT  = obsB + (size_t)NTOK * OD_DIM;
  __bf16* WT   = oWT + (size_t)OD_DIM * C_DIM;

  dim3 blk(128);
  dim3 gProj(C_DIM / 64, NTOK / 128);      // 8 x 64
  dim3 gFF1(FF_DIM / 64, NTOK / 128);      // 32 x 64
  dim3 gAttn(T_DIM / 64, B_DIM * H_DIM);   // 16 x 64
  dim3 gTcc(C_DIM / 32, C_DIM / 32);

  // ---- prep: bf16 embedding input + transposed bf16 weights ----
  cvt_bf16_kernel<<<(NTOK * OD_DIM) / 1024, 256, 0, stream>>>(obs, obsB,
                                                              NTOK * OD_DIM);
  transpose_cvt_bf16<<<dim3(C_DIM / 32, OD_DIM / 32), 256, 0, stream>>>(
      obs_W, oWT, OD_DIM, C_DIM);
  for (int i = 0; i < 6; ++i) {
    __bf16* lw = WT + (size_t)i * LWSEG;
    const size_t wo  = (size_t)i * CC;
    const size_t w1o = (size_t)i * CF;
    transpose_cvt_bf16<<<gTcc, 256, 0, stream>>>(Wq + wo, lw + 0 * CC, C_DIM, C_DIM);
    transpose_cvt_bf16<<<gTcc, 256, 0, stream>>>(Wk + wo, lw + 1 * CC, C_DIM, C_DIM);
    transpose_cvt_bf16<<<gTcc, 256, 0, stream>>>(Wv + wo, lw + 2 * CC, C_DIM, C_DIM);
    transpose_cvt_bf16<<<gTcc, 256, 0, stream>>>(Wo + wo, lw + 3 * CC, C_DIM, C_DIM);
    transpose_cvt_bf16<<<dim3(FF_DIM / 32, C_DIM / 32), 256, 0, stream>>>(
        W1 + w1o, lw + 4 * CC, C_DIM, FF_DIM);
    transpose_cvt_bf16<<<dim3(C_DIM / 32, FF_DIM / 32), 256, 0, stream>>>(
        W2 + w1o, lw + 4 * CC + CF, FF_DIM, C_DIM);
  }

  // ---- embedding: x = gelu(obs @ obs_W + obs_b) + pos_emb[:T] ----
  gemm_bf16_async<3><<<gProj, blk, 0, stream>>>(obsB, oWT, obs_b, pos_emb, X,
                                                NTOK, C_DIM, OD_DIM);

  for (int i = 0; i < 6; ++i) {
    __bf16* lw  = WT + (size_t)i * LWSEG;
    __bf16* WqT = lw + 0 * CC;
    __bf16* WkT = lw + 1 * CC;
    __bf16* WvT = lw + 2 * CC;
    __bf16* WoT = lw + 3 * CC;
    __bf16* W1T = lw + 4 * CC;
    __bf16* W2T = lw + 4 * CC + CF;

    layernorm_512<1><<<NTOK, 256, 0, stream>>>(X, ln1_w + i * C_DIM,
                                               ln1_b + i * C_DIM, Hb);
    gemm_bf16_async<0><<<gProj, blk, 0, stream>>>(Hb, WqT, bq + i * C_DIM,
                                                  nullptr, Qb, NTOK, C_DIM, C_DIM);
    gemm_bf16_async<0><<<gProj, blk, 0, stream>>>(Hb, WkT, bk + i * C_DIM,
                                                  nullptr, Kb, NTOK, C_DIM, C_DIM);
    gemm_bf16_async<4><<<gProj, blk, 0, stream>>>(Hb, WvT, bv + i * C_DIM,
                                                  nullptr, Vtb, NTOK, C_DIM, C_DIM);
    attn_flash<<<gAttn, blk, 0, stream>>>(Qb, Kb, Vtb, Yb);
    gemm_bf16_async<2><<<gProj, blk, 0, stream>>>(Yb, WoT, bo + i * C_DIM, X, X,
                                                  NTOK, C_DIM, C_DIM);
    layernorm_512<1><<<NTOK, 256, 0, stream>>>(X, ln2_w + i * C_DIM,
                                               ln2_b + i * C_DIM, Hb);
    gemm_bf16_async<1><<<gFF1, blk, 0, stream>>>(Hb, W1T,
                                                 b1 + (size_t)i * FF_DIM,
                                                 nullptr, HID, NTOK, FF_DIM, C_DIM);
    gemm_bf16_async<2><<<gProj, blk, 0, stream>>>(HID, W2T, b2 + i * C_DIM, X, X,
                                                  NTOK, C_DIM, FF_DIM);
  }
  layernorm_512<0><<<NTOK, 256, 0, stream>>>(X, lnf_w, lnf_b, d_out);
}
